// EGNNLayer_9088150798461
// MI455X (gfx1250) — matrix operand order repaired and checked by reading, not compiled, not run on hardware
//
#include <hip/hip_runtime.h>
#include <hip/hip_bf16.h>

typedef __attribute__((ext_vector_type(16))) _Float16 v16h;
typedef __attribute__((ext_vector_type(8)))  _Float16 v8h;
typedef __attribute__((ext_vector_type(8)))  float    v8f;
typedef __attribute__((ext_vector_type(4)))  float    v4f;

#define H64 64
#define DE16 16
#define KPAD 160   // 2H+1+DE = 145 padded to 160 (5 k-tiles of 32)

__global__ void egnn_init_kernel(float* __restrict__ agg,
                                 float* __restrict__ pos_out,
                                 const float* __restrict__ pos,
                                 int n64, int n3) {
    int i = blockIdx.x * blockDim.x + threadIdx.x;
    if (i < n64) agg[i] = 0.0f;
    if (i < n3)  pos_out[i] = pos[i];
}

__global__ __launch_bounds__(256) void egnn_edge_kernel(
    const float* __restrict__ h, const float* __restrict__ pos,
    const int* __restrict__ ei, const float* __restrict__ attr,
    const float* __restrict__ We1, const float* __restrict__ be1,
    const float* __restrict__ We2, const float* __restrict__ be2,
    const float* __restrict__ Wc1, const float* __restrict__ bc1,
    const float* __restrict__ Wc2, const float* __restrict__ bc2,
    float* __restrict__ agg, float* __restrict__ pos_out,
    int N, int E) {

    // Transposed f16 weights in LDS: Wt[out][in], in-contiguous so A-fragments
    // are two 16B ds_load_b128 per (m-tile, k-tile).
    __shared__ __align__(16) _Float16 sWe1t[H64 * KPAD];   // 20 KB
    __shared__ __align__(16) _Float16 sWe2t[H64 * H64];    // 8 KB
    __shared__ __align__(16) _Float16 sWc1t[H64 * H64];    // 8 KB
    __shared__ float sWc2[H64];
    __shared__ float sBe1[H64], sBe2[H64], sBc1[H64];
    __shared__ float sBc2v;
    // per-wave activation staging, [edge][feature] f16, 2KB per wave
    __shared__ __align__(16) _Float16 sM[8 * 16 * H64];    // 16 KB

    const int tid = threadIdx.x;
    for (int i = tid; i < H64 * KPAD; i += 256) {
        int o = i / KPAD, in = i % KPAD;
        sWe1t[i] = (in < 145) ? (_Float16)We1[in * H64 + o] : (_Float16)0.0f;
    }
    for (int i = tid; i < H64 * H64; i += 256) {
        int o = i >> 6, in = i & 63;
        sWe2t[i] = (_Float16)We2[in * H64 + o];
        sWc1t[i] = (_Float16)Wc1[in * H64 + o];
    }
    if (tid < H64) {
        sWc2[tid] = Wc2[tid];
        sBe1[tid] = be1[tid]; sBe2[tid] = be2[tid]; sBc1[tid] = bc1[tid];
    }
    if (tid == 0) sBc2v = bc2[0];
    __syncthreads();

    const int wave  = tid >> 5;
    const int lane  = tid & 31;
    const int erow  = lane & 15;   // edge within tile (B/D column)
    const int khalf = lane >> 4;   // which K half-group this lane carries
    _Float16* mrow = &sM[wave * (16 * H64) + erow * H64];

    const int gwave  = blockIdx.x * 8 + wave;
    const int stride = gridDim.x * 8;
    const int ntiles = (E + 15) >> 4;

    // 64->64 f16 WMMA layer: B fragments from mrow staging, A from Wt in LDS.
    auto layer64 = [&](const _Float16* Wt, const float* bias, v8f* out) {
        #pragma unroll
        for (int mt = 0; mt < 4; ++mt)
            #pragma unroll
            for (int i = 0; i < 8; ++i) out[mt][i] = bias[mt * 16 + khalf * 8 + i];
        #pragma unroll
        for (int kt = 0; kt < 2; ++kt) {
            const _Float16* bp = mrow + kt * 32 + khalf * 16;
            v8h b0 = *(const v8h*)(bp);
            v8h b1 = *(const v8h*)(bp + 8);
            v16h B = __builtin_shufflevector(b0, b1, 0,1,2,3,4,5,6,7,8,9,10,11,12,13,14,15);
            #pragma unroll
            for (int mt = 0; mt < 4; ++mt) {
                const _Float16* wr = &Wt[(mt * 16 + erow) * H64 + kt * 32 + khalf * 16];
                v8h a0 = *(const v8h*)(wr);
                v8h a1 = *(const v8h*)(wr + 8);
                v16h A = __builtin_shufflevector(a0, a1, 0,1,2,3,4,5,6,7,8,9,10,11,12,13,14,15);
                out[mt] = __builtin_amdgcn_wmma_f32_16x16x32_f16(
                    false, A, false, B, (short)0, out[mt], false, false);
            }
        }
    };

    for (int tile = gwave; tile < ntiles; tile += stride) {
        const int e  = (tile << 4) + erow;
        const bool ev = (e < E);
        const int ec = ev ? e : (E - 1);
        const int s = ei[ec];
        const int d = ei[E + ec];

        const float dx = pos[d * 3 + 0] - pos[s * 3 + 0];
        const float dy = pos[d * 3 + 1] - pos[s * 3 + 1];
        const float dz = pos[d * 3 + 2] - pos[s * 3 + 2];
        const float d2  = fminf(dx * dx + dy * dy + dz * dz, 1000.0f);
        const float inv = 1.0f / sqrtf(d2 + 1e-8f);

        const float* hs = h + (size_t)s * H64;
        const float* hd = h + (size_t)d * H64;
        const float* ar = attr + (size_t)ec * DE16;

        // ---- edge MLP layer 1: [16 x 160] input tile x Wt -> 64 features ----
        v8f acc[4];
        #pragma unroll
        for (int mt = 0; mt < 4; ++mt)
            #pragma unroll
            for (int i = 0; i < 8; ++i) acc[mt][i] = sBe1[mt * 16 + khalf * 8 + i];

        #pragma unroll
        for (int kt = 0; kt < 5; ++kt) {
            v16h B;
            if (kt < 4) {  // columns 0..127: h[src] then h[dst], contiguous 16-runs
                const float* bp = (kt < 2 ? hs : hd) + (kt & 1) * 32 + khalf * 16;
                v4f f0 = *(const v4f*)(bp + 0);
                v4f f1 = *(const v4f*)(bp + 4);
                v4f f2 = *(const v4f*)(bp + 8);
                v4f f3 = *(const v4f*)(bp + 12);
                #pragma unroll
                for (int j = 0; j < 4; ++j) {
                    B[j]      = (_Float16)f0[j];
                    B[4 + j]  = (_Float16)f1[j];
                    B[8 + j]  = (_Float16)f2[j];
                    B[12 + j] = (_Float16)f3[j];
                }
            } else {       // columns 128..159: [dist2, attr[0..15], zeros]
                if (khalf == 0) {
                    v4f a0 = *(const v4f*)(ar + 0);
                    v4f a1 = *(const v4f*)(ar + 4);
                    v4f a2 = *(const v4f*)(ar + 8);
                    v4f a3 = *(const v4f*)(ar + 12);
                    B[0] = (_Float16)d2;
                    #pragma unroll
                    for (int j = 0; j < 4; ++j) {
                        B[1 + j] = (_Float16)a0[j];
                        B[5 + j] = (_Float16)a1[j];
                        B[9 + j] = (_Float16)a2[j];
                    }
                    B[13] = (_Float16)a3[0]; B[14] = (_Float16)a3[1]; B[15] = (_Float16)a3[2];
                } else {
                    B[0] = (_Float16)ar[15];
                    #pragma unroll
                    for (int j = 1; j < 16; ++j) B[j] = (_Float16)0.0f;
                }
            }
            #pragma unroll
            for (int mt = 0; mt < 4; ++mt) {
                const _Float16* wr = &sWe1t[(mt * 16 + erow) * KPAD + kt * 32 + khalf * 16];
                v8h a0 = *(const v8h*)(wr);
                v8h a1 = *(const v8h*)(wr + 8);
                v16h A = __builtin_shufflevector(a0, a1, 0,1,2,3,4,5,6,7,8,9,10,11,12,13,14,15);
                acc[mt] = __builtin_amdgcn_wmma_f32_16x16x32_f16(
                    false, A, false, B, (short)0, acc[mt], false, false);
            }
        }

        // relu -> stage f16 activations [edge][feature] for next layer's B frags
        #pragma unroll
        for (int mt = 0; mt < 4; ++mt) {
            v8h p;
            #pragma unroll
            for (int i = 0; i < 8; ++i) p[i] = (_Float16)fmaxf(acc[mt][i], 0.0f);
            *(v8h*)(mrow + mt * 16 + khalf * 8) = p;
        }
        asm volatile("s_wait_dscnt 0" ::: "memory");

        // ---- edge MLP layer 2 -> m_ij ----
        v8f m2[4];
        layer64(sWe2t, sBe2, m2);

        float* aggr = agg + (size_t)d * H64;
        #pragma unroll
        for (int mt = 0; mt < 4; ++mt) {
            v8h p;
            #pragma unroll
            for (int i = 0; i < 8; ++i) {
                float r = fmaxf(m2[mt][i], 0.0f);
                p[i] = (_Float16)r;
                if (ev) atomicAdd(&aggr[mt * 16 + khalf * 8 + i], r);  // scatter m_ij
            }
            *(v8h*)(mrow + mt * 16 + khalf * 8) = p;
        }
        asm volatile("s_wait_dscnt 0" ::: "memory");

        // ---- coord MLP layer 1 ----
        v8f c3[4];
        layer64(sWc1t, sBc1, c3);

        // ---- coord MLP layer 2 (64 -> 1): VALU dot + cross-half shuffle ----
        float part = 0.0f;
        #pragma unroll
        for (int mt = 0; mt < 4; ++mt)
            #pragma unroll
            for (int i = 0; i < 8; ++i)
                part += fmaxf(c3[mt][i], 0.0f) * sWc2[mt * 16 + khalf * 8 + i];
        float other = __shfl_xor(part, 16, 32);
        float coef = tanhf(part + other + sBc2v) * 0.1f;

        if ((lane < 16) && ev) {
            atomicAdd(&pos_out[(size_t)d * 3 + 0], dx * inv * coef);
            atomicAdd(&pos_out[(size_t)d * 3 + 1], dy * inv * coef);
            atomicAdd(&pos_out[(size_t)d * 3 + 2], dz * inv * coef);
        }
    }
}

__global__ __launch_bounds__(256) void egnn_node_kernel(
    const float* __restrict__ h, const float* __restrict__ agg,
    const float* __restrict__ Wn, const float* __restrict__ bn,
    const float* __restrict__ lng, const float* __restrict__ lnb,
    float* __restrict__ hout, int N) {

    __shared__ float sWn[128 * H64];   // 32 KB
    __shared__ float sbn[H64];
    __shared__ float sIn[8][128];

    const int tid = threadIdx.x;
    for (int i = tid; i < 128 * H64; i += 256) sWn[i] = Wn[i];
    if (tid < H64) sbn[tid] = bn[tid];
    __syncthreads();

    const int wave = tid >> 5;
    const int lane = tid & 31;
    float* in = sIn[wave];

    for (int n = blockIdx.x * 8 + wave; n < N; n += gridDim.x * 8) {
        const float h0 = h[(size_t)n * H64 + lane];
        const float h1 = h[(size_t)n * H64 + 32 + lane];
        in[lane]      = h0;
        in[32 + lane] = h1;
        in[64 + lane] = agg[(size_t)n * H64 + lane];
        in[96 + lane] = agg[(size_t)n * H64 + 32 + lane];
        asm volatile("s_wait_dscnt 0" ::: "memory");

        float a0 = sbn[lane], a1 = sbn[32 + lane];
        #pragma unroll 8
        for (int k = 0; k < 128; ++k) {
            const float x = in[k];
            a0 += x * sWn[k * H64 + lane];
            a1 += x * sWn[k * H64 + 32 + lane];
        }
        const float x0 = h0 + fmaxf(a0, 0.0f);
        const float x1 = h1 + fmaxf(a1, 0.0f);

        float ssum = x0 + x1;
        #pragma unroll
        for (int o = 16; o > 0; o >>= 1) ssum += __shfl_xor(ssum, o, 32);
        const float mu = ssum * (1.0f / 64.0f);
        const float d0 = x0 - mu, d1 = x1 - mu;
        float vsum = d0 * d0 + d1 * d1;
        #pragma unroll
        for (int o = 16; o > 0; o >>= 1) vsum += __shfl_xor(vsum, o, 32);
        const float rinv = 1.0f / sqrtf(vsum * (1.0f / 64.0f) + 1e-5f);

        hout[(size_t)n * H64 + lane]      = d0 * rinv * lng[lane] + lnb[lane];
        hout[(size_t)n * H64 + 32 + lane] = d1 * rinv * lng[32 + lane] + lnb[32 + lane];
        asm volatile("s_wait_dscnt 0" ::: "memory");  // staging reuse across iterations
    }
}

extern "C" void kernel_launch(void* const* d_in, const int* in_sizes, int n_in,
                              void* d_out, int out_size, void* d_ws, size_t ws_size,
                              hipStream_t stream) {
    (void)n_in; (void)out_size; (void)ws_size;
    const float* h    = (const float*)d_in[0];
    const float* pos  = (const float*)d_in[1];
    const int*   ei   = (const int*)d_in[2];
    const float* attr = (const float*)d_in[3];
    const float* We1  = (const float*)d_in[4];
    const float* be1  = (const float*)d_in[5];
    const float* We2  = (const float*)d_in[6];
    const float* be2  = (const float*)d_in[7];
    const float* Wc1  = (const float*)d_in[8];
    const float* bc1  = (const float*)d_in[9];
    const float* Wc2  = (const float*)d_in[10];
    const float* bc2  = (const float*)d_in[11];
    const float* Wn   = (const float*)d_in[12];
    const float* bn   = (const float*)d_in[13];
    const float* lng  = (const float*)d_in[14];
    const float* lnb  = (const float*)d_in[15];

    const int N = in_sizes[1] / 3;
    const int E = in_sizes[2] / 2;

    float* hout = (float*)d_out;
    float* pout = hout + (size_t)N * H64;   // output tuple: h_out then pos_out
    float* agg  = (float*)d_ws;             // [N,64] f32 scatter target (L2-resident)

    const int n64 = N * H64;
    egnn_init_kernel<<<(n64 + 255) / 256, 256, 0, stream>>>(agg, pout, pos, n64, N * 3);

    const int ntiles = (E + 15) >> 4;
    int eblocks = (ntiles + 7) / 8;
    if (eblocks > 2048) eblocks = 2048;
    if (eblocks < 1) eblocks = 1;
    egnn_edge_kernel<<<eblocks, 256, 0, stream>>>(
        h, pos, ei, attr, We1, be1, We2, be2, Wc1, bc1, Wc2, bc2,
        agg, pout, N, E);

    int nblocks = (N + 7) / 8;
    if (nblocks > 8192) nblocks = 8192;
    if (nblocks < 1) nblocks = 1;
    egnn_node_kernel<<<nblocks, 256, 0, stream>>>(h, agg, Wn, bn, lng, lnb, hout, N);
}
